// NodeFormerConv_137438953628
// MI455X (gfx1250) — compile-verified, matrix operand order checked by reading, not run on hardware
//
#include <hip/hip_runtime.h>
#include <hip/hip_bf16.h>
#include <math.h>

typedef __attribute__((ext_vector_type(16))) __bf16 v16bf;
typedef __attribute__((ext_vector_type(8)))  float  v8f;

#define CC 256   // in channels = H*D
#define HH 4     // heads
#define DD 64    // head dim (also out channels)
#define MM 30    // projection rows
#define MP 32    // padded M
// (1/sqrt(TAU)) * D^-0.25 = 2 * 64^-0.25 = 2^-0.5
#define FSCALE 0.70710678118654752440f
// M^-0.5 = 30^-0.5
#define RATIO  0.18257418583505537115f
#define EPSV   1e-6f

__device__ __forceinline__ void atomicMaxF(float* a, float v) {
  if (v >= 0.f) atomicMax((int*)a, __float_as_int(v));
  else          atomicMin((unsigned int*)a, __float_as_uint(v));
}

// Load a 16x32 bf16 A (or 32x16 B) fragment. Matrix stored "major-major":
// element (major, k) at p[major*ld + k]. Per ISA 16-bit layout:
// lane = hi*16 + major ; VGPR j holds K pair (j<4 ? 2j : 16+2(j-4)) + 8*hi.
__device__ __forceinline__ v16bf frag16(const __bf16* p, int ld, int major, int kbase, int hi) {
  union { v16bf v; unsigned u[8]; } r;
  const __bf16* b = p + (size_t)major * ld + kbase + hi * 8;
  r.u[0] = *(const unsigned*)(b + 0);
  r.u[1] = *(const unsigned*)(b + 2);
  r.u[2] = *(const unsigned*)(b + 4);
  r.u[3] = *(const unsigned*)(b + 6);
  r.u[4] = *(const unsigned*)(b + 16);
  r.u[5] = *(const unsigned*)(b + 18);
  r.u[6] = *(const unsigned*)(b + 20);
  r.u[7] = *(const unsigned*)(b + 22);
  return r.v;
}

__device__ __forceinline__ v8f wmma_bf16(v16bf a, v16bf b, v8f c) {
  return __builtin_amdgcn_wmma_f32_16x16x32_bf16(false, a, false, b, (short)0, c, false, false);
}

// ---------------------------------------------------------------- init
__global__ void k_init(float* degin, float* degout, float* ug, float* og,
                       float* lossb, float* kmax, float* sgm, const float* brb, int N) {
  int idx = blockIdx.x * blockDim.x + threadIdx.x;
  int stride = gridDim.x * blockDim.x;
  for (int i = idx; i < N; i += stride) { degin[i] = 0.f; degout[i] = 0.f; }
  for (int i = idx; i < HH * MM * DD; i += stride) ug[i] = 0.f;
  if (idx < HH * MM) og[idx] = 0.f;
  if (idx == 0) lossb[0] = 0.f;
  if (idx < HH) {
    kmax[idx] = -INFINITY;
    sgm[idx] = 1.f / (1.f + __expf(-brb[idx]));
  }
}

// ------------------------------------------------- weight conversion (bf16, transposed)
__global__ void k_convert(const float* Wq, const float* Wk, const float* Wv,
                          const float* Wo, const float* proj,
                          __bf16* WqT, __bf16* WkT, __bf16* WvT, __bf16* WoT, __bf16* projP) {
  int i = blockIdx.x * blockDim.x + threadIdx.x;
  if (i < CC * CC) {
    int n = i / CC, k = i % CC;                 // WT[out][in] = W[in][out]
    WqT[i] = (__bf16)Wq[k * CC + n];
    WkT[i] = (__bf16)Wk[k * CC + n];
    WvT[i] = (__bf16)Wv[k * CC + n];
  }
  if (i < DD * CC) {
    int n = i / CC, k = i % CC;
    WoT[i] = (__bf16)Wo[k * DD + n];
  }
  if (i < MP * DD) {
    int n = i / DD, k = i % DD;                 // projP[m][d], rows >= 30 zero
    projP[i] = (n < MM) ? (__bf16)proj[n * DD + k] : (__bf16)0.f;
  }
}

// ---------------------------------------------------------------- fused pass 1
// 16 nodes per block, 128 threads (4 waves). WMMA QKV, then per-head WMMA dash,
// diag, per-node qmax -> qp store; dash_k - diag_k store + global kmax atomic.
__global__ void __launch_bounds__(128) k_pass1(
    const float* z, const float* bq, const float* bk, const float* bv,
    const __bf16* WqT, const __bf16* WkT, const __bf16* WvT, const __bf16* projP,
    float* vbuf, float* qp, float* dkb, float* kmaxg, int N) {
  __shared__ __bf16 zt[16][CC];
  __shared__ __bf16 qs[16][CC];
  __shared__ __bf16 kt[16][CC];
  __shared__ float dq[HH][16][MP];
  __shared__ float dk[HH][16][MP];
  __shared__ float dgq[16][HH], dgk[16][HH], kml[16][HH];
  const int t = threadIdx.x;
  const int lane = t & 31, wv = t >> 5;
  const int nl = lane & 15, hi = lane >> 4;
  const long nb = (long)blockIdx.x * 16;

  for (int i = t; i < 16 * CC; i += 128)
    zt[i >> 8][i & 255] = (__bf16)z[nb * CC + i];
  __syncthreads();

  // ---- QKV: wave wv does coltiles {wv, wv+4, wv+8, wv+12}
  for (int c4 = 0; c4 < 4; ++c4) {
    int ct = wv + c4 * 4;
    int cb = ct * 16;
    v8f aq = {}, ak = {}, av = {};
    for (int ksi = 0; ksi < 8; ++ksi) {
      int kb = ksi * 32;
      v16bf af  = frag16(&zt[0][0], CC, nl, kb, hi);
      v16bf bqf = frag16(WqT, CC, cb + nl, kb, hi);
      v16bf bkf = frag16(WkT, CC, cb + nl, kb, hi);
      v16bf bvf = frag16(WvT, CC, cb + nl, kb, hi);
      aq = wmma_bf16(af, bqf, aq);
      ak = wmma_bf16(af, bkf, ak);
      av = wmma_bf16(af, bvf, av);
    }
    int col = cb + nl;
#pragma unroll
    for (int r = 0; r < 8; ++r) {
      int M = r + 8 * hi;
      float qv = (aq[r] + bq[col]) * FSCALE;
      float kv = (ak[r] + bk[col]) * FSCALE;
      qs[M][col] = (__bf16)qv;
      kt[M][col] = (__bf16)kv;
      vbuf[(nb + M) * CC + col] = av[r] + bv[col];
    }
  }
  __syncthreads();

  // ---- dash: wave wv = head wv, output 16 nodes x 32 (padded M)
  {
    int h = wv;
    for (int n0 = 0; n0 < 2; ++n0) {
      v8f cq = {}, ck = {};
      for (int ksi = 0; ksi < 2; ++ksi) {
        int kb = ksi * 32;
        v16bf aqf = frag16(&qs[0][0], CC, nl, h * DD + kb, hi);
        v16bf akf = frag16(&kt[0][0], CC, nl, h * DD + kb, hi);
        v16bf bpf = frag16(projP, DD, n0 * 16 + nl, kb, hi);
        cq = wmma_bf16(aqf, bpf, cq);
        ck = wmma_bf16(akf, bpf, ck);
      }
#pragma unroll
      for (int r = 0; r < 8; ++r) {
        int M = r + 8 * hi;
        dq[h][M][n0 * 16 + nl] = cq[r];
        dk[h][M][n0 * 16 + nl] = ck[r];
      }
    }
  }
  __syncthreads();

  // ---- diag = 0.5*sum(data^2) per (node, head)
  if (t < 64) {
    int n = t >> 2, h = t & 3;
    float s = 0.f;
    for (int d = 0; d < DD; ++d) { float x = (float)qs[n][h * DD + d]; s += x * x; }
    dgq[n][h] = 0.5f * s;
  } else {
    int tt = t - 64; int n = tt >> 2, h = tt & 3;
    float s = 0.f;
    for (int d = 0; d < DD; ++d) { float x = (float)kt[n][h * DD + d]; s += x * x; }
    dgk[n][h] = 0.5f * s;
  }
  __syncthreads();

  if (t < 64) {  // q: per-node stab -> qp
    int n = t >> 2, h = t & 3;
    float mx = -INFINITY;
    for (int m = 0; m < MM; ++m) mx = fmaxf(mx, dq[h][n][m]);
    float dg = dgq[n][h];
    float* out = qp + ((nb + n) * HH + h) * MP;
    for (int m = 0; m < MM; ++m)
      out[m] = RATIO * (__expf(dq[h][n][m] - dg - mx) + EPSV);
    out[MM] = 0.f; out[MM + 1] = 0.f;
  } else {       // k: store dash-diag, collect tile max
    int tt = t - 64; int n = tt >> 2, h = tt & 3;
    float mx = -INFINITY;
    float dg = dgk[n][h];
    float* out = dkb + ((nb + n) * HH + h) * MP;
    for (int m = 0; m < MM; ++m) {
      float dv = dk[h][n][m];
      mx = fmaxf(mx, dv);
      out[m] = dv - dg;
    }
    kml[n][h] = mx;
  }
  __syncthreads();
  if (t < HH) {
    float mx = -INFINITY;
    for (int n = 0; n < 16; ++n) mx = fmaxf(mx, kml[n][t]);
    atomicMaxF(&kmaxg[t], mx);
  }
}

// ------------------------- kp = ratio*(exp(dash-diag-kmax)+eps); u_k, o_k reduction
__global__ void __launch_bounds__(256) k_kpu(float* kpbuf, const float* vbuf,
                                             const float* kmaxg, float* ug, float* og, int N) {
  __shared__ float kpL[HH][MM];
  const int t = threadIdx.x;
  const int th = t / MM, tm = t % MM;           // valid for t < 120
  const int h = t >> 6, d = t & 63;
  int chunk = (N + gridDim.x - 1) / gridDim.x;
  long n0 = (long)blockIdx.x * chunk;
  long n1 = n0 + chunk; if (n1 > N) n1 = N;
  float km = (t < HH * MM) ? kmaxg[th] : 0.f;
  float oreg = 0.f;
  float ureg[MM];
#pragma unroll
  for (int m = 0; m < MM; ++m) ureg[m] = 0.f;

  for (long n = n0; n < n1; ++n) {
    if (t < HH * MM) {
      long ix = (n * HH + th) * MP + tm;
      float kv = RATIO * (__expf(kpbuf[ix] - km) + EPSV);
      kpbuf[ix] = kv;            // overwrite dash buffer with kp
      kpL[th][tm] = kv;
      oreg += kv;
    } else if (t < HH * MM + 8) {
      int p = t - HH * MM;       // zero the 2 pad slots of each head
      kpbuf[(n * HH + (p >> 1)) * MP + MM + (p & 1)] = 0.f;
    }
    __syncthreads();
    float vv = vbuf[n * CC + t]; // t == h*64+d
#pragma unroll
    for (int m = 0; m < MM; ++m) ureg[m] += kpL[h][m] * vv;
    __syncthreads();
  }
#pragma unroll
  for (int m = 0; m < MM; ++m) atomicAdd(&ug[(h * MM + m) * DD + d], ureg[m]);
  if (t < HH * MM) atomicAdd(&og[th * MM + tm], oreg);
}

// ------------------------------------------------ degrees
__global__ void k_deg(const int* ei, float* degin, float* degout, int E) {
  int e = blockIdx.x * blockDim.x + threadIdx.x;
  if (e < E) {
    atomicAdd(&degout[ei[e]], 1.f);       // row
    atomicAdd(&degin[ei[E + e]], 1.f);    // col
  }
}

// ---------------------------------- z_num, z_den, z_out = num/den
__global__ void __launch_bounds__(256) k_znum(const float* qp, const float* ug, const float* og,
                                              float* zout, float* zden, int N) {
  __shared__ float uL[HH * MM * DD];
  __shared__ float oL[HH * MM];
  __shared__ float qL[HH][MM];
  __shared__ float zdL[HH];
  const int t = threadIdx.x;
  for (int i = t; i < HH * MM * DD; i += 256) uL[i] = ug[i];
  if (t < HH * MM) oL[t] = og[t];
  __syncthreads();
  const int h = t >> 6, d = t & 63;
  int chunk = (N + gridDim.x - 1) / gridDim.x;
  long n0 = (long)blockIdx.x * chunk;
  long n1 = n0 + chunk; if (n1 > N) n1 = N;
  for (long n = n0; n < n1; ++n) {
    if (t < HH * MM) qL[t / MM][t % MM] = qp[(n * HH + t / MM) * MP + (t % MM)];
    __syncthreads();
    if (t < HH) {
      float s = 0.f;
      for (int m = 0; m < MM; ++m) s += qL[t][m] * oL[t * MM + m];
      zdL[t] = s;
      zden[n * HH + t] = s;
    }
    __syncthreads();
    float acc = 0.f;
#pragma unroll
    for (int m = 0; m < MM; ++m) acc += qL[h][m] * uL[(h * MM + m) * DD + d];
    zout[n * CC + t] = acc / zdL[h];
    __syncthreads();
  }
}

// ------------------------------- edge attention + link loss
__global__ void __launch_bounds__(256) k_att(const int* ei, const float* qp, const float* kp,
                                             const float* zden, const float* degin,
                                             float* lossb, int E) {
  __shared__ float red[256];
  int e = blockIdx.x * blockDim.x + threadIdx.x;
  float ls = 0.f;
  if (e < E) {
    int r = ei[e], c = ei[E + e];
    float lsum = 0.f;
    for (int h = 0; h < HH; ++h) {
      const float4* a = (const float4*)(qp + ((long)c * HH + h) * MP);
      const float4* b = (const float4*)(kp + ((long)r * HH + h) * MP);
      float s = 0.f;
#pragma unroll
      for (int j = 0; j < 8; ++j) {   // pads are zero in both -> safe
        float4 x = a[j], y = b[j];
        s += x.x * y.x + x.y * y.y + x.z * y.z + x.w * y.w;
      }
      lsum += __logf(s / zden[(long)c * HH + h]);
    }
    ls = lsum / degin[c];
  }
  red[threadIdx.x] = ls;
  __syncthreads();
  for (int s = 128; s > 0; s >>= 1) {
    if (threadIdx.x < s) red[threadIdx.x] += red[threadIdx.x + s];
    __syncthreads();
  }
  if (threadIdx.x == 0) atomicAdd(lossb, red[0]);
}

// ------------------------------- relational-bias scatter: zout[row] += val_h * v[col]
__global__ void __launch_bounds__(64) k_bias(const int* ei, const float* degin, const float* degout,
                                             const float* sgm, const float* vbuf,
                                             float* zout, int E) {
  int e = blockIdx.x;
  if (e >= E) return;
  int r = ei[e], c = ei[E + e];
  float sc = rsqrtf(degin[c]) * rsqrtf(degout[r]);
  int t = threadIdx.x;
#pragma unroll
  for (int j = 0; j < 4; ++j) {       // j == head
    int idx = t + j * 64;
    float vv = vbuf[(long)c * CC + idx];
    atomicAdd(&zout[(long)r * CC + idx], sgm[j] * sc * vv);
  }
}

// ------------------------------- output GEMM: z_next = zout @ Wo + bo (WMMA)
__global__ void __launch_bounds__(128) k_out(const float* zout, const __bf16* WoT,
                                             const float* bo, float* out, int N) {
  __shared__ __bf16 zt[16][CC];
  const int t = threadIdx.x;
  const int lane = t & 31, wv = t >> 5;
  const int nl = lane & 15, hi = lane >> 4;
  const long nb = (long)blockIdx.x * 16;
  for (int i = t; i < 16 * CC; i += 128)
    zt[i >> 8][i & 255] = (__bf16)zout[nb * CC + i];
  __syncthreads();
  int cb = wv * 16;
  v8f acc = {};
  for (int ksi = 0; ksi < 8; ++ksi) {
    v16bf a = frag16(&zt[0][0], CC, nl, ksi * 32, hi);
    v16bf b = frag16(WoT, CC, cb + nl, ksi * 32, hi);
    acc = wmma_bf16(a, b, acc);
  }
  int col = cb + nl;
#pragma unroll
  for (int r = 0; r < 8; ++r) {
    int M = r + 8 * hi;
    out[(nb + M) * DD + col] = acc[r] + bo[col];
  }
}

__global__ void k_fin(const float* lossb, float* out, long pos, float inv) {
  if (threadIdx.x == 0 && blockIdx.x == 0) out[pos] = lossb[0] * inv;
}

// ================================================================ launch
extern "C" void kernel_launch(void* const* d_in, const int* in_sizes, int n_in,
                              void* d_out, int out_size, void* d_ws, size_t ws_size,
                              hipStream_t stream) {
  const float* z    = (const float*)d_in[0];
  const int*   ei   = (const int*)d_in[1];
  const float* Wq   = (const float*)d_in[2];
  const float* bq   = (const float*)d_in[3];
  const float* Wk   = (const float*)d_in[4];
  const float* bk   = (const float*)d_in[5];
  const float* Wv   = (const float*)d_in[6];
  const float* bv   = (const float*)d_in[7];
  const float* Wo   = (const float*)d_in[8];
  const float* bo   = (const float*)d_in[9];
  const float* brb  = (const float*)d_in[10];
  const float* proj = (const float*)d_in[11];
  const int N = in_sizes[0] / CC;
  const int E = in_sizes[1] / 2;

  float* w = (float*)d_ws;
  size_t off = 0;
  float* qp     = w + off; off += (size_t)N * HH * MP;
  float* kp     = w + off; off += (size_t)N * HH * MP;   // dash_k first, kp after k_kpu
  float* vbuf   = w + off; off += (size_t)N * CC;
  float* zout   = w + off; off += (size_t)N * CC;
  float* zden   = w + off; off += (size_t)N * HH;
  float* degin  = w + off; off += N;
  float* degout = w + off; off += N;
  float* ug     = w + off; off += HH * MM * DD;
  float* og     = w + off; off += HH * MM;
  float* kmax   = w + off; off += HH;
  float* lossb  = w + off; off += 1;
  float* sgm    = w + off; off += 4;
  __bf16* WqT   = (__bf16*)(w + off);
  __bf16* WkT   = WqT + CC * CC;
  __bf16* WvT   = WkT + CC * CC;
  __bf16* WoT   = WvT + CC * CC;
  __bf16* projP = WoT + DD * CC;
  float* outz = (float*)d_out;

  k_init<<<512, 256, 0, stream>>>(degin, degout, ug, og, lossb, kmax, sgm, brb, N);
  k_convert<<<(CC * CC + 255) / 256, 256, 0, stream>>>(Wq, Wk, Wv, Wo, proj,
                                                       WqT, WkT, WvT, WoT, projP);
  k_pass1<<<N / 16, 128, 0, stream>>>(z, bq, bk, bv, WqT, WkT, WvT, projP,
                                      vbuf, qp, kp, kmax, N);
  k_kpu<<<256, 256, 0, stream>>>(kp, vbuf, kmax, ug, og, N);
  k_deg<<<(E + 255) / 256, 256, 0, stream>>>(ei, degin, degout, E);
  k_znum<<<256, 256, 0, stream>>>(qp, ug, og, zout, zden, N);
  k_att<<<(E + 255) / 256, 256, 0, stream>>>(ei, qp, kp, zden, degin, lossb, E);
  k_bias<<<E, 64, 0, stream>>>(ei, degin, degout, sgm, vbuf, zout, E);
  k_out<<<N / 16, 128, 0, stream>>>(zout, WoT, bo, outz, N);
  k_fin<<<1, 1, 0, stream>>>(lossb, outz, (long)N * DD, 1.0f / ((float)E * HH));
}
